// HalutMatmul_75196287418368
// MI455X (gfx1250) — matrix operand order repaired and checked by reading, not compiled, not run on hardware
//
#include <hip/hip_runtime.h>
#include <hip/hip_bf16.h>

// HalutMatmul on gfx1250 (wave32, WGP):
//   K1: P = I@A via v_wmma_f32_16x16x32_f16 + fused depth-4 tree descent -> idx u8 (ws)
//   K2: out = onehot(idx) @ L^T. One-hot E is always 2:4-expressible (<=1 nonzero per
//       group of 4 K) -> v_swmmac_f32_16x16x64_f16 (K=64/instr), wmma fallback.
// All WMMA operands are loaded as contiguous ds_load_b128 pairs (transposed packed LDS).

typedef __attribute__((ext_vector_type(16))) _Float16 v16h;
typedef __attribute__((ext_vector_type(32))) _Float16 v32h;
typedef __attribute__((ext_vector_type(8)))  float    v8f;
typedef __attribute__((ext_vector_type(8)))  int      v8i;

#define HN 32768
#define HD 512
#define HC 64
#define HM 512
#define HCD 256   // C*DEPTH

struct alignas(16) Frag8  { uint4 lo, hi; };            // 32B -> v16h
struct alignas(16) Frag16 { uint4 q0, q1, q2, q3; };    // 64B -> v32h

__device__ __forceinline__ unsigned pack2h(float x, float y) {
  _Float16 hx = (_Float16)x, hy = (_Float16)y;
  unsigned short ux = __builtin_bit_cast(unsigned short, hx);
  unsigned short uy = __builtin_bit_cast(unsigned short, hy);
  return (unsigned)ux | ((unsigned)uy << 16);
}

// ---------------- Kernel 1: P = I@A (f16 WMMA) + tree descent -> idx ----------------
// Block: 256 thr (8 waves), Ntile = 128 rows, all 256 features.
// LDS: Ipk u32[128][20] (row n x d-pair), Apk u32[256][20] (col f x d-pair),
//      Plds f32[128][264], Tlds f32[960]
#define K1_IPK   (128 * 20)
#define K1_APK   (256 * 20)
#define K1_PLDS  (128 * 264)
#define K1_SMEM  ((K1_IPK + K1_APK + K1_PLDS + 960) * 4)

__global__ __launch_bounds__(256) void halut_k1(const float* __restrict__ I,
                                                const float* __restrict__ T,
                                                const float* __restrict__ A,
                                                unsigned char* __restrict__ idx) {
  extern __shared__ char smem[];
  unsigned* Ipk  = (unsigned*)smem;
  unsigned* Apk  = Ipk + K1_IPK;
  float*    Plds = (float*)(Apk + K1_APK);
  float*    Tlds = Plds + K1_PLDS;

  const int tid  = threadIdx.x;
  const int n0   = blockIdx.x * 128;
  const int w    = tid >> 5;
  const int lane = tid & 31;
  const int li   = lane & 15;
  const int h    = lane >> 4;

  for (int i = tid; i < 960; i += 256) Tlds[i] = T[i];

  v8f acc[16];
  #pragma unroll
  for (int j = 0; j < 16; ++j) acc[j] = (v8f)0.0f;

  for (int dc = 0; dc < HD; dc += 32) {
    __syncthreads();
    // Ipk[nn][p] = pack(I[n0+nn][dc+2p], I[n0+nn][dc+2p+1])
    for (int e = tid; e < 2048; e += 256) {
      int nn = e >> 4, p = e & 15;
      const float2 v = *reinterpret_cast<const float2*>(I + (size_t)(n0 + nn) * HD + dc + 2 * p);
      Ipk[nn * 20 + p] = pack2h(v.x, v.y);
    }
    // Apk[f][q] = pack(A[dc+2q][f], A[dc+2q+1][f])
    for (int e = tid; e < 4096; e += 256) {
      int q = e >> 8, f = e & 255;
      float x = A[(size_t)(dc + 2 * q) * HCD + f];
      float y = A[(size_t)(dc + 2 * q + 1) * HCD + f];
      Apk[f * 20 + q] = pack2h(x, y);
    }
    __syncthreads();

    // A fragment: vgpr r -> pair (r&3)+4h+8(r>>2): two contiguous b128 runs
    Frag8 fa;
    fa.lo = *reinterpret_cast<const uint4*>(&Ipk[(w * 16 + li) * 20 + 4 * h]);
    fa.hi = *reinterpret_cast<const uint4*>(&Ipk[(w * 16 + li) * 20 + 8 + 4 * h]);
    v16h a = __builtin_bit_cast(v16h, fa);

    #pragma unroll
    for (int j = 0; j < 16; ++j) {
      // B fragment: vgpr r -> pair (r&3)+4h+8(r>>2): two contiguous b128 runs
      Frag8 fb;
      fb.lo = *reinterpret_cast<const uint4*>(&Apk[(j * 16 + li) * 20 + 4 * h]);
      fb.hi = *reinterpret_cast<const uint4*>(&Apk[(j * 16 + li) * 20 + 8 + 4 * h]);
      v16h b = __builtin_bit_cast(v16h, fb);
      acc[j] = __builtin_amdgcn_wmma_f32_16x16x32_f16(false, a, false, b,
                                                      (short)0, acc[j], false, false);
    }
  }

  // spill P tile per C/D layout: (vgpr r, lane) -> row r+8h, col li
  #pragma unroll
  for (int j = 0; j < 16; ++j)
    #pragma unroll
    for (int r = 0; r < 8; ++r)
      Plds[(w * 16 + r + 8 * h) * 264 + j * 16 + li] = acc[j][r];
  __syncthreads();

  // tree descent: 128n x 64c cells
  for (int e = tid; e < 8192; e += 256) {
    int nn = e >> 6, c = e & 63;
    const float4 p4 = *reinterpret_cast<const float4*>(&Plds[nn * 264 + 4 * c]);
    float pv[4] = {p4.x, p4.y, p4.z, p4.w};
    int node = 0, k = 0;
    #pragma unroll
    for (int l = 0; l < 4; ++l) {
      int bit = (pv[l] > Tlds[c * 15 + node]) ? 1 : 0;
      k = (k << 1) | bit;
      node = 2 * node + 1 + bit;
    }
    idx[(size_t)(n0 + nn) * HC + c] = (unsigned char)k;
  }
}

// ---------------- Kernel 2: out = onehot(idx) @ L^T ----------------
// Block: 256 thr (8 waves), tile = 128n x 128m, K = 1024 (64 codebooks x 16).
// LDS: Lpk u32[128][516] (m x packed f16 k-pair), idx u8[128][64]
#define K2_LPK  (128 * 516)
#define K2_SMEM (K2_LPK * 4 + 128 * 64)

__global__ __launch_bounds__(256) void halut_k2(const unsigned char* __restrict__ idx,
                                                const float* __restrict__ L,
                                                float* __restrict__ out) {
  extern __shared__ char smem[];
  unsigned*      Lpk    = (unsigned*)smem;           // [128][516]
  unsigned char* idxLds = (unsigned char*)(Lpk + K2_LPK);

  const int tid = threadIdx.x;
  const int n0  = blockIdx.x * 128;
  const int m0  = blockIdx.y * 128;

  // stage L slice transposed-packed: Lpk[ml][pg] = pack(L[m0+ml][2pg], L[m0+ml][2pg+1])
  {
    int ml = tid & 127, hs = tid >> 7;
    const float* Lrow = L + (size_t)(m0 + ml) * 1024;
    for (int i = 0; i < 256; ++i) {
      int pg = hs * 256 + i;
      const float2 v = *reinterpret_cast<const float2*>(Lrow + 2 * pg);
      Lpk[ml * 516 + pg] = pack2h(v.x, v.y);
    }
  }
  // stage idx tile (128x64 u8)
  {
    const unsigned* src = reinterpret_cast<const unsigned*>(idx + (size_t)n0 * HC);
    unsigned* dst = reinterpret_cast<unsigned*>(idxLds);
    for (int e = tid; e < 2048; e += 256) dst[e] = src[e];
  }
  __syncthreads();

  const int w    = tid >> 5;
  const int lane = tid & 31;
  const int li   = lane & 15;
  const int h    = lane >> 4;

  v8f acc[8];
  #pragma unroll
  for (int j = 0; j < 8; ++j) acc[j] = (v8f)0.0f;

#if defined(__AMDGCN__) && __has_builtin(__builtin_amdgcn_swmmac_f32_16x16x64_f16)
  // -------- SWMMAC path: 2:4 sparse one-hot A, K=64 per instruction --------
  for (int kc = 0; kc < 16; ++kc) {   // 64-wide K chunk = codebooks 4kc..4kc+3
    unsigned iv = *reinterpret_cast<const unsigned*>(idxLds + (w * 16 + li) * HC + 4 * kc);
    int gk[4], pp[4];
    #pragma unroll
    for (int c = 0; c < 4; ++c) {
      int b = (iv >> (8 * c)) & 255;
      gk[c] = b >> 2;      // group-of-4 holding the nonzero
      pp[c] = b & 3;       // position within group
    }
    // index VGPR (ISA 7.12.4): lane half h holds groups 8h+j (j=0..7), 4 bits each
    unsigned ireg = 0;
    #pragma unroll
    for (int j = 0; j < 8; ++j) {
      int cc = 2 * h + (j >> 2);
      int lg = j & 3;
      unsigned code = (gk[cc] == lg)
                          ? (unsigned)(((pp[cc] < 3) ? pp[cc] : 2) | 12)  // (idx0, 3)
                          : 4u;                                           // (0, 1)
      ireg |= code << (4 * j);
    }
    // stored A (16x32 layout): vgpr r -> group (r&3)+4h+8(r>>2) -> cb h+2(r>>2), lg r&3
    v8i ai;
    #pragma unroll
    for (int r = 0; r < 8; ++r) {
      int cc = h + 2 * (r >> 2);
      int lg = r & 3;
      bool match = (gk[cc] == lg);
      unsigned v = 0;
      v |= (match && pp[cc] < 3) ? 0x3C00u : 0u;        // slot0 = value at idx0
      v |= (match && pp[cc] == 3) ? 0x3C000000u : 0u;   // slot1 = value at idx1(=3)
      ai[r] = (int)v;
    }
    v16h a = __builtin_bit_cast(v16h, ai);

    #pragma unroll
    for (int j = 0; j < 8; ++j) {
      // dense B 64x16: vgpr r<8 -> pair r+8h; r>=8 -> pair 16+(r-8)+8h : 4x b128
      const unsigned* base = &Lpk[(size_t)(j * 16 + li) * 516 + kc * 32 + 8 * h];
      Frag16 fb;
      fb.q0 = *reinterpret_cast<const uint4*>(base + 0);
      fb.q1 = *reinterpret_cast<const uint4*>(base + 4);
      fb.q2 = *reinterpret_cast<const uint4*>(base + 16);
      fb.q3 = *reinterpret_cast<const uint4*>(base + 20);
      v32h b = __builtin_bit_cast(v32h, fb);
      // prototype (probe-derived): (neg_a, A, neg_b, B, C, index, reuse_a, reuse_b)
      acc[j] = __builtin_amdgcn_swmmac_f32_16x16x64_f16(false, a, false, b,
                                                        acc[j], (int)ireg, false, false);
    }
  }
#else
  // -------- WMMA fallback: dense one-hot A, K=32 per instruction --------
  for (int kc = 0; kc < 32; ++kc) {   // 32-wide K chunk = codebooks 2kc, 2kc+1
    unsigned short pr =
        *reinterpret_cast<const unsigned short*>(idxLds + (w * 16 + li) * HC + 2 * kc);
    int i0 = pr & 0xFF, i1 = pr >> 8;
    v8i ai;
    #pragma unroll
    for (int r = 0; r < 8; ++r) {
      int kb  = 2 * (r & 3) + 8 * h;
      int sel = (r < 4) ? i0 : i1;
      unsigned v = ((sel == kb) ? 0x3C00u : 0u) | ((sel == kb + 1) ? 0x3C000000u : 0u);
      ai[r] = (int)v;
    }
    v16h a = __builtin_bit_cast(v16h, ai);
    #pragma unroll
    for (int j = 0; j < 8; ++j) {
      const unsigned* base = &Lpk[(size_t)(j * 16 + li) * 516 + kc * 16 + 4 * h];
      Frag8 fb;
      fb.lo = *reinterpret_cast<const uint4*>(base + 0);
      fb.hi = *reinterpret_cast<const uint4*>(base + 8);
      v16h b = __builtin_bit_cast(v16h, fb);
      acc[j] = __builtin_amdgcn_wmma_f32_16x16x32_f16(false, a, false, b,
                                                      (short)0, acc[j], false, false);
    }
  }
#endif

  // write out per C/D layout: (vgpr r, lane) -> n = w*16 + r + 8h, m = j*16 + li
  #pragma unroll
  for (int j = 0; j < 8; ++j)
    #pragma unroll
    for (int r = 0; r < 8; ++r) {
      int n = n0 + w * 16 + r + 8 * h;
      int m = m0 + j * 16 + li;
      out[(size_t)n * HM + m] = acc[j][r];
    }
}

extern "C" void kernel_launch(void* const* d_in, const int* in_sizes, int n_in,
                              void* d_out, int out_size, void* d_ws, size_t ws_size,
                              hipStream_t stream) {
  const float* I = (const float*)d_in[0];
  const float* T = (const float*)d_in[1];
  const float* L = (const float*)d_in[2];
  const float* A = (const float*)d_in[3];
  // d_in[4] = S, d_in[5] = B: structure baked into the tree-descent kernel.
  unsigned char* idx = (unsigned char*)d_ws;  // N*C = 2 MB, fully rewritten each call
  float* out = (float*)d_out;

  (void)in_sizes; (void)n_in; (void)out_size; (void)ws_size;

  (void)hipFuncSetAttribute(reinterpret_cast<const void*>(halut_k1),
                            hipFuncAttributeMaxDynamicSharedMemorySize, K1_SMEM);
  (void)hipFuncSetAttribute(reinterpret_cast<const void*>(halut_k2),
                            hipFuncAttributeMaxDynamicSharedMemorySize, K2_SMEM);

  halut_k1<<<dim3(HN / 128), 256, K1_SMEM, stream>>>(I, T, A, idx);
  halut_k2<<<dim3(HN / 128, HM / 128), 256, K2_SMEM, stream>>>(idx, L, out);
}
